// MultiQueryAttention_64278480552137
// MI455X (gfx1250) — matrix operand order repaired and checked by reading, not compiled, not run on hardware
//
#include <hip/hip_runtime.h>

typedef __attribute__((ext_vector_type(16))) _Float16 v16h;
typedef __attribute__((ext_vector_type(8)))  _Float16 h8;
typedef __attribute__((ext_vector_type(8)))  float    v8f;
typedef unsigned int u32x4 __attribute__((ext_vector_type(4)));
typedef int          i32x8 __attribute__((ext_vector_type(8)));
typedef int          i32x4 __attribute__((ext_vector_type(4)));

union V16 { v16h v; h8 h[2]; };

constexpr int Dm   = 1024;
constexpr int Hn   = 16;
constexpr int HD   = 64;
constexpr int Bsz  = 2;
constexpr int Sl   = 2048;
constexpr int MROWS = Bsz * Sl;   // 4096

__device__ inline v8f wmma_f16(v16h a, v16h b, v8f c) {
  // D = A(16x32 f16) * B(32x16 f16) + C(16x16 f32)
  return __builtin_amdgcn_wmma_f32_16x16x32_f16(false, a, false, b, (short)0, c,
                                                false, false);
}

__global__ void cvt_f32_to_f16(const float* __restrict__ src,
                               _Float16* __restrict__ dst, int n) {
  int i = blockIdx.x * blockDim.x + threadIdx.x;
  if (i < n) dst[i] = (_Float16)src[i];
}

// C[M,N] = A[M,K](f16) @ W[N,K]^T(f16), f32 accumulate.
// One wave computes a 32x64 tile: 8 WMMAs per 32-wide K step, each B tile
// reused by two A tiles. No manual pipelining — the scheduler forms load
// clauses and unrolls on its own (manual double-buffering caused spills).
__global__ void gemm_xwt(const _Float16* __restrict__ A,
                         const _Float16* __restrict__ W,
                         float* __restrict__ Cf, _Float16* __restrict__ Ch,
                         int M, int N, int K) {
  int lane  = threadIdx.x & 31;
  int wave  = (blockIdx.x * blockDim.x + threadIdx.x) >> 5;
  int nb    = N >> 6;
  int mi    = (wave / nb) << 5;     // 32 rows per wave
  int n0    = (wave % nb) << 6;     // 64 cols per wave
  if (mi >= M) return;
  int lhalf = lane >> 4;            // 0/1
  int l15   = lane & 15;

  v8f acc[2][4] = {};
  const _Float16* arow0 = A + (size_t)(mi + l15) * K + lhalf * 8;
  const _Float16* arow1 = arow0 + (size_t)16 * K;
  const _Float16* brow  = W + (size_t)(n0 + l15) * K + lhalf * 8;

  for (int kc = 0; kc < K; kc += 32) {
    __builtin_prefetch(arow0 + kc + 512, 0, 1);      // global_prefetch_b8
    __builtin_prefetch(brow + kc + 512, 0, 1);
    V16 a0, a1;
    a0.h[0] = *(const h8*)(arow0 + kc);        // K: 0-7 / 8-15
    a0.h[1] = *(const h8*)(arow0 + kc + 16);   // K: 16-23 / 24-31
    a1.h[0] = *(const h8*)(arow1 + kc);
    a1.h[1] = *(const h8*)(arow1 + kc + 16);
#pragma unroll
    for (int t = 0; t < 4; ++t) {
      const _Float16* p = brow + (size_t)(t * 16) * K + kc;
      V16 b;
      b.h[0] = *(const h8*)(p);                // K: 0-7 / 8-15
      b.h[1] = *(const h8*)(p + 16);           // K: 16-23 / 24-31
      acc[0][t] = wmma_f16(a0.v, b.v, acc[0][t]);
      acc[1][t] = wmma_f16(a1.v, b.v, acc[1][t]);
    }
  }

  int mrow = lhalf * 8;
#pragma unroll
  for (int half = 0; half < 2; ++half) {
#pragma unroll
    for (int t = 0; t < 4; ++t) {
#pragma unroll
      for (int r = 0; r < 8; ++r) {
        size_t idx = (size_t)(mi + half * 16 + mrow + r) * N + n0 + t * 16 + l15;
        if (Cf) Cf[idx] = acc[half][t][r];
        if (Ch) Ch[idx] = (_Float16)acc[half][t][r];
      }
    }
  }
}

// Flash-attention: one wave per (b, head, 16-query tile); single shared KV head.
__global__ void mqa_attn(const _Float16* __restrict__ Q,
                         const _Float16* __restrict__ Kc,
                         const _Float16* __restrict__ Vc,
                         _Float16* __restrict__ Ah) {
  __shared__ __align__(16) float    lds_s[16][32];
  __shared__ __align__(16) _Float16 lds_p[16][32];
  __shared__ __align__(16) _Float16 lds_v[32][64];
  __shared__ float lds_fac[16];
  __shared__ float lds_rs[16];

  int lane  = threadIdx.x & 31;
  int l15   = lane & 15;
  int lhalf = lane >> 4;
  int mrow  = lhalf * 8;

  int tile = blockIdx.x;
  int s0   = (tile & (Sl / 16 - 1)) * 16;
  int hh   = (tile / (Sl / 16)) & (Hn - 1);
  int b    = tile / ((Sl / 16) * Hn);

  // Q tile in A-layout: row = l15, split K 0..31 / 32..63.
  const _Float16* qrow = Q + (size_t)(b * Sl + s0 + l15) * Dm + hh * HD + lhalf * 8;
  V16 qa0, qa1;
  qa0.h[0] = *(const h8*)(qrow);        qa0.h[1] = *(const h8*)(qrow + 16);
  qa1.h[0] = *(const h8*)(qrow + 32);   qa1.h[1] = *(const h8*)(qrow + 48);

  const float scale = 0.125f;                 // 1/sqrt(64)
  const float L2E   = 1.4426950408889634f;
  float rmax = -1e30f, rsum = 0.0f;           // row stats (valid in lanes 0..15)
  v8f o[4] = {};

  int nch = (s0 + 16 + 31) >> 5;              // causal: only needed 32-key chunks
  for (int c = 0; c < nch; ++c) {
    int t0 = c * 32;

    // ---- stage V chunk (32 keys x 64 dims, 4KB) into LDS via TDM ----
#if __has_builtin(__builtin_amdgcn_tensor_load_to_lds)
    {
      // make sure previous chunk's ds reads of lds_v have drained
      asm volatile("s_wait_dscnt 0x0" ::: "memory");
      unsigned lds_base = (unsigned)(uintptr_t)(&lds_v[0][0]);
      unsigned long long ga =
          (unsigned long long)(uintptr_t)(Vc + (size_t)(b * Sl + t0) * HD);
      // D# group0: count=1, lds_addr, global_addr, type=2
      u32x4 g0 = { 1u, lds_base, (unsigned)ga,
                   (unsigned)((ga >> 32) & 0x01FFFFFFu) | 0x80000000u };
      // D# group1: data_size=2B, tensor_dim0=64, tensor_dim1=rows left
      // (TDM zero-fills OOB rows; their P columns are 0 anyway),
      // tile_dim0=64, tile_dim1=32, tensor_dim0_stride=64
      int dim1 = Sl - t0;
      i32x8 g1 = { (int)0x00010000,
                   (int)(64u << 16),
                   (int)((unsigned)dim1 << 16),
                   (int)(64u << 16),
                   32, 64, 0, 0 };
      i32x4 g2 = {0, 0, 0, 0};
      i32x4 g3 = {0, 0, 0, 0};
#if __clang_major__ >= 23
      i32x8 g4 = {0, 0, 0, 0, 0, 0, 0, 0};
      __builtin_amdgcn_tensor_load_to_lds(g0, g1, g2, g3, g4, 0);
#else
      __builtin_amdgcn_tensor_load_to_lds(g0, g1, g2, g3, 0);
#endif
    }
#else
#pragma unroll
    for (int i = 0; i < 8; ++i) {
      int e   = i * 32 + lane;
      int row = e >> 3;
      int co  = (e & 7) * 8;
      int key = t0 + row; if (key > Sl - 1) key = Sl - 1;
      *(h8*)(&lds_v[row][co]) = *(const h8*)(Vc + (size_t)(b * Sl + key) * HD + co);
    }
#endif

    // ---- scores: Qtile(16x64) @ K^T -> two 16x16 f32 tiles (32 keys) ----
    v8f sc[2] = {};
#pragma unroll
    for (int half = 0; half < 2; ++half) {
      int key = t0 + half * 16 + l15; if (key > Sl - 1) key = Sl - 1;  // masked later
      const _Float16* krow = Kc + (size_t)(b * Sl + key) * HD + lhalf * 8;
      V16 b0, b1;
      b0.h[0] = *(const h8*)(krow);        b0.h[1] = *(const h8*)(krow + 16);
      b1.h[0] = *(const h8*)(krow + 32);   b1.h[1] = *(const h8*)(krow + 48);
      sc[half] = wmma_f16(qa0.v, b0.v, sc[half]);
      sc[half] = wmma_f16(qa1.v, b1.v, sc[half]);
    }
    __builtin_prefetch(Kc + (size_t)(b * Sl + t0 + 32 + lane) * HD, 0, 1);

    // ---- scale + causal mask, scatter C-layout scores to LDS rows ----
#pragma unroll
    for (int r = 0; r < 8; ++r) {
      int qi = s0 + mrow + r;
      float x0 = sc[0][r] * scale;
      float x1 = sc[1][r] * scale;
      if (t0 + l15 > qi)      x0 = -1e30f;
      if (t0 + 16 + l15 > qi) x1 = -1e30f;
      lds_s[mrow + r][l15]      = x0;
      lds_s[mrow + r][16 + l15] = x1;
    }
    __syncthreads();

    // ---- online softmax row reduction (lane m owns row m) ----
    if (lane < 16) {
      float mx = rmax;
#pragma unroll
      for (int j = 0; j < 32; ++j) mx = fmaxf(mx, lds_s[lane][j]);
      float fac = __builtin_amdgcn_exp2f((rmax - mx) * L2E);
      float sum = 0.0f;
#pragma unroll
      for (int j = 0; j < 32; ++j) {
        float p = __builtin_amdgcn_exp2f((lds_s[lane][j] - mx) * L2E);
        sum += p;
        lds_p[lane][j] = (_Float16)p;
      }
      rsum = rsum * fac + sum;
      rmax = mx;
      lds_fac[lane] = fac;
    }
    __syncthreads();

    // ---- rescale running O by per-row factor ----
#pragma unroll
    for (int r = 0; r < 8; ++r) {
      float f = lds_fac[mrow + r];
      o[0][r] *= f; o[1][r] *= f; o[2][r] *= f; o[3][r] *= f;
    }

    // ---- wait for TDM V tile, then O += P(16x32) @ V(32x64) ----
#if __has_builtin(__builtin_amdgcn_tensor_load_to_lds)
#if __has_builtin(__builtin_amdgcn_s_wait_tensorcnt)
    __builtin_amdgcn_s_wait_tensorcnt(0);
#else
    asm volatile("s_wait_tensorcnt 0x0" ::: "memory");
#endif
#endif
    V16 pa;
    pa.h[0] = *(const h8*)(&lds_p[l15][lhalf * 8]);
    pa.h[1] = *(const h8*)(&lds_p[l15][16 + lhalf * 8]);
#pragma unroll
    for (int t = 0; t < 4; ++t) {
      v16h vb;
#pragma unroll
      for (int kk = 0; kk < 16; ++kk)
        vb[kk] = lds_v[(kk & 7) + ((kk >> 3) << 4) + lhalf * 8][t * 16 + l15];   // B-layout gather
      o[t] = wmma_f16(pa.v, vb, o[t]);
    }
    __syncthreads();
  }

  if (lane < 16) lds_rs[lane] = rsum;
  __syncthreads();
#pragma unroll
  for (int r = 0; r < 8; ++r) {
    int m = mrow + r;
    float inv = 1.0f / lds_rs[m];
    size_t row = (size_t)(b * Sl + s0 + m) * Dm + hh * HD;
#pragma unroll
    for (int t = 0; t < 4; ++t)
      Ah[row + t * 16 + l15] = (_Float16)(o[t][r] * inv);
  }
}

extern "C" void kernel_launch(void* const* d_in, const int* in_sizes, int n_in,
                              void* d_out, int out_size, void* d_ws, size_t ws_size,
                              hipStream_t stream) {
  const float* x  = (const float*)d_in[0];
  const float* Wq = (const float*)d_in[1];
  const float* Wk = (const float*)d_in[2];
  const float* Wv = (const float*)d_in[3];
  const float* Wo = (const float*)d_in[4];
  // d_in[5] = mask: causal, applied analytically in-kernel.
  float* out = (float*)d_out;

  _Float16* w   = (_Float16*)d_ws;
  _Float16* xh  = w;
  _Float16* Wqh = xh  + (size_t)MROWS * Dm;
  _Float16* Wkh = Wqh + (size_t)Dm * Dm;
  _Float16* Wvh = Wkh + (size_t)HD * Dm;
  _Float16* Woh = Wvh + (size_t)HD * Dm;
  _Float16* Qh  = Woh + (size_t)Dm * Dm;
  _Float16* Kh  = Qh  + (size_t)MROWS * Dm;
  _Float16* Vh  = Kh  + (size_t)MROWS * HD;
  _Float16* Ah  = Vh  + (size_t)MROWS * HD;   // total ~30.7 MB of ws

  auto cvt = [&](const float* s, _Float16* d, int n) {
    cvt_f32_to_f16<<<(n + 255) / 256, 256, 0, stream>>>(s, d, n);
  };
  cvt(x,  xh,  MROWS * Dm);
  cvt(Wq, Wqh, Dm * Dm);
  cvt(Wk, Wkh, HD * Dm);
  cvt(Wv, Wvh, HD * Dm);
  cvt(Wo, Woh, Dm * Dm);

  // Q = x @ Wq^T  (f16 out)
  {
    int waves = (MROWS / 32) * (Dm / 64);         // 2048
    gemm_xwt<<<waves * 32 / 256, 256, 0, stream>>>(xh, Wqh, nullptr, Qh,
                                                   MROWS, Dm, Dm);
  }
  // K = x @ Wk^T ; V = x @ Wv^T  (f16 out)
  {
    int waves = (MROWS / 32) * (HD / 64);         // 128
    gemm_xwt<<<waves * 32 / 256, 256, 0, stream>>>(xh, Wkh, nullptr, Kh,
                                                   MROWS, HD, Dm);
    gemm_xwt<<<waves * 32 / 256, 256, 0, stream>>>(xh, Wvh, nullptr, Vh,
                                                   MROWS, HD, Dm);
  }
  // flash attention: B*H*(S/16) = 4096 waves
  mqa_attn<<<Bsz * Hn * (Sl / 16), 32, 0, stream>>>(Qh, Kh, Vh, Ah);
  // out = attn @ Wo^T  (f32 out)
  {
    int waves = (MROWS / 32) * (Dm / 64);
    gemm_xwt<<<waves * 32 / 256, 256, 0, stream>>>(Ah, Woh, out, nullptr,
                                                   MROWS, Dm, Dm);
  }
}
